// Self_Attention_26895085208060
// MI455X (gfx1250) — compile-verified, matrix operand order checked by reading, not run on hardware
//
#include <hip/hip_runtime.h>
#include <hip/hip_bf16.h>

// ---------------- CDNA5 types ----------------
typedef __attribute__((ext_vector_type(16))) __bf16   v16bf;
typedef __attribute__((ext_vector_type(8)))  float    v8f;
typedef __attribute__((ext_vector_type(4)))  float    v4f;
typedef __attribute__((ext_vector_type(4)))  unsigned v4u;
typedef __attribute__((ext_vector_type(4)))  int      v4i;
typedef __attribute__((ext_vector_type(8)))  int      v8i;

#define BB 4
#define TT 2048
#define CC 1024
#define NEG_INF (-__builtin_inff())

// f32 -> bf16 RNE via bit ops
static __device__ __forceinline__ __bf16 f2bf(float f) {
    union { float f; unsigned u; } in; in.f = f;
    unsigned u = in.u;
    u += 0x7fffu + ((u >> 16) & 1u);
    unsigned short h = (unsigned short)(u >> 16);
    union { unsigned short s; __bf16 b; } out; out.s = h;
    return out.b;
}

// 16-bit fragment for V_WMMA_*_16X16X32 (CDNA5 ISA layout):
// lane L: row/col = L%16; K pairs: VGPR r<4 -> K=(L>=16?8:0)+2r; r>=4 -> 16+(L>=16?8:0)+2(r-4)
union Frag { v16bf v; unsigned u[8]; };

static __device__ __forceinline__ void load_frag(Frag& f, const __bf16* __restrict__ p, int ld) {
    const int lane = threadIdx.x & 31;
    const __bf16* base = p + (size_t)(lane & 15) * ld + ((lane >> 4) << 3);
#pragma unroll
    for (int i = 0; i < 4; ++i) {
        f.u[i]     = *(const unsigned*)(base + 2 * i);
        f.u[i + 4] = *(const unsigned*)(base + 16 + 2 * i);
    }
}

// B fragment out of the LDS-staged tile: row-major [64 cols(n)][32 k], ld = 32
static __device__ __forceinline__ void load_frag_lds(Frag& f, const __bf16* tile, int nLocal) {
    const int lane = threadIdx.x & 31;
    const __bf16* base = tile + (size_t)(nLocal + (lane & 15)) * 32 + ((lane >> 4) << 3);
#pragma unroll
    for (int i = 0; i < 4; ++i) {
        f.u[i]     = *(const unsigned*)(base + 2 * i);
        f.u[i + 4] = *(const unsigned*)(base + 16 + 2 * i);
    }
}

// ---------------- Tensor Data Mover: async 2-D tile load into LDS ----------------
#if __has_builtin(__builtin_amdgcn_tensor_load_to_lds)
#define HAVE_TDM 1
#else
#define HAVE_TDM 0
#endif

static __device__ __forceinline__ void tdm_load_tile_2d(unsigned lds_off, const __bf16* gsrc,
                                                        unsigned tensor_d0, unsigned tensor_d1,
                                                        long long stride0_elems,
                                                        unsigned tile_d0, unsigned tile_d1) {
#if HAVE_TDM
    const unsigned long long ga = (unsigned long long)(size_t)gsrc;
    v4u g0;
    g0[0] = 1u;                                                // count=1, user descriptor
    g0[1] = lds_off;                                           // lds_addr (bytes)
    g0[2] = (unsigned)(ga & 0xffffffffu);                      // global_addr[31:0]
    g0[3] = (unsigned)((ga >> 32) & 0x01ffffffu) | (2u << 30); // global_addr[56:32], type=2
    v8i g1;
    g1[0] = (int)(1u << 16);                                   // wg_mask=0, data_size=1 (2B)
    g1[1] = (int)((tensor_d0 & 0xffffu) << 16);                // tensor_dim0[15:0]
    g1[2] = (int)((tensor_d0 >> 16) | ((tensor_d1 & 0xffffu) << 16));
    g1[3] = (int)((tensor_d1 >> 16) | (tile_d0 << 16));        // tile_dim0
    g1[4] = (int)(tile_d1 & 0xffffu);                          // tile_dim1 (tile_dim2=0)
    g1[5] = (int)(unsigned)(stride0_elems & 0xffffffffll);     // tensor_dim0_stride[31:0]
    g1[6] = (int)(unsigned)((stride0_elems >> 32) & 0xffffll); // stride[47:32]
    g1[7] = 0;
    v4i z4 = {0, 0, 0, 0};
#if defined(__clang_major__) && (__clang_major__ >= 23)
    v8i z8 = {0, 0, 0, 0, 0, 0, 0, 0};
    __builtin_amdgcn_tensor_load_to_lds(g0, g1, z4, z4, z8, 0);
#else
    __builtin_amdgcn_tensor_load_to_lds(g0, g1, z4, z4, 0);
#endif
#else
    (void)lds_off; (void)gsrc; (void)tensor_d0; (void)tensor_d1;
    (void)stride0_elems; (void)tile_d0; (void)tile_d1;
#endif
}

static __device__ __forceinline__ void tdm_wait_le1() {
#if HAVE_TDM
    __builtin_amdgcn_s_wait_tensorcnt((short)1);
#endif
}
static __device__ __forceinline__ void tdm_wait_le0() {
#if HAVE_TDM
    __builtin_amdgcn_s_wait_tensorcnt((short)0);
#endif
}

// ---------------- elementwise f32 -> bf16 (4 elems / thread) ----------------
__global__ void cvt_f32_to_bf16_k(const float* __restrict__ in, __bf16* __restrict__ out, int n) {
    const int i = (blockIdx.x * blockDim.x + threadIdx.x) * 4;
    if (i < n) {                 // n is a multiple of 4 in all uses
        v4f v = *(const v4f*)(in + i);
        union { __bf16 b[4]; unsigned long long q; } pk;
#pragma unroll
        for (int j = 0; j < 4; ++j) pk.b[j] = f2bf(v[j]);
        *(unsigned long long*)(out + i) = pk.q;
    }
}

// ---------------- LDS-tiled transpose of V: qkv[b][t][2C+c] -> vt[b][c][t] ----------------
__global__ void transpose_v_k(const __bf16* __restrict__ qkvb, __bf16* __restrict__ vt) {
    __shared__ __bf16 tile[32][33];
    const int b  = blockIdx.z;
    const int t0 = blockIdx.x * 32;
    const int c0 = blockIdx.y * 32;
    const int tx = threadIdx.x, ty = threadIdx.y;
    tile[ty][tx] = qkvb[((size_t)b * TT + (t0 + ty)) * (3 * CC) + 2 * CC + c0 + tx];
    __syncthreads();
    vt[((size_t)b * CC + (c0 + ty)) * TT + t0 + tx] = tile[tx][ty];
}

// ---------------- unified WMMA GEMM: C = A[M,K] * Bt[N,K]^T ----------------
// Block: 8 waves stacked in M (tile 128x64). Wave tile: 16x64 (A frag reused over 4 acc).
// B tile (64x32 bf16) double-buffer staged into LDS by wave 7 via TDM, overlapped with
// WMMA; A fragments software-pipelined (global prefetch one k-step ahead) so the
// pre-WMMA wait only covers ds loads.
__global__ void gemm_bf16_wmma_k(const __bf16* __restrict__ A, const __bf16* __restrict__ Bt,
                                 void* __restrict__ Cout,
                                 int M, int N, int K, int lda, int ldb, int ldc,
                                 int mode, float scale, const float* __restrict__ bias,
                                 long long sA, long long sB, long long sC, int causal) {
    __shared__ __bf16 tileB[2][64 * 32];   // 2 x 4 KB

    const int bz = blockIdx.z;
    A  += (size_t)bz * sA;
    Bt += (size_t)bz * sB;

    const int lane = threadIdx.x & 31;
    const int wave = threadIdx.x >> 5;
    const int m0 = blockIdx.x * 128 + wave * 16;
    const int n0 = blockIdx.y * 64;
    if (m0 >= M) return;

    const int nl = lane & 15;
    const int mb = (lane >> 4) * 8;

    if (causal && mode == 1 && n0 > m0 + 15) {
        // wave tile strictly above diagonal: emit -inf and terminate (barriers count
        // terminated waves as arrived, CDNA5 ISA)
        float* Cf = (float*)Cout + (size_t)bz * sC;
#pragma unroll
        for (int nt = 0; nt < 4; ++nt)
#pragma unroll
            for (int r = 0; r < 8; ++r)
                Cf[(size_t)(m0 + mb + r) * ldc + (n0 + nt * 16 + nl)] = NEG_INF;
        return;
    }

    // wave 7 owns data movement: largest m0 in block -> if it is causally masked,
    // every other wave is too.
    const bool issuer = (wave == 7);
    const int  kIters = K / 32;           // always even here (32 or 64)

    if (issuer) {
#if HAVE_TDM
        tdm_load_tile_2d((unsigned)(size_t)&tileB[0][0], Bt + (size_t)n0 * ldb,
                         (unsigned)K, (unsigned)(N - n0), ldb, 32u, 64u);
#else
        for (int r = lane; r < 64; r += 32) {
            const v4u* s = (const v4u*)(Bt + (size_t)(n0 + r) * ldb);
            v4u* d = (v4u*)&tileB[0][r * 32];
            d[0] = s[0]; d[1] = s[1]; d[2] = s[2]; d[3] = s[3];
        }
#endif
    }

    v8f acc[4] = {};
    const __bf16* aRow = A + (size_t)m0 * lda;

    Frag a0, a1;
    load_frag(a0, aRow, lda);             // A prefetch for step 0

    auto kstep = [&](int i, Frag& aCur, Frag& aNxt) {
        const int cur = i & 1;
        if (issuer) {
            if (i + 1 < kIters) {
#if HAVE_TDM
                tdm_load_tile_2d((unsigned)(size_t)&tileB[cur ^ 1][0],
                                 Bt + (size_t)n0 * ldb + (i + 1) * 32,
                                 (unsigned)(K - (i + 1) * 32), (unsigned)(N - n0),
                                 ldb, 32u, 64u);
#else
                for (int r = lane; r < 64; r += 32) {
                    const v4u* s = (const v4u*)(Bt + (size_t)(n0 + r) * ldb + (i + 1) * 32);
                    v4u* d = (v4u*)&tileB[cur ^ 1][r * 32];
                    d[0] = s[0]; d[1] = s[1]; d[2] = s[2]; d[3] = s[3];
                }
#endif
                tdm_wait_le1();           // buffer `cur` complete (TDM in-order per wave)
            } else {
                tdm_wait_le0();
            }
        }
        __syncthreads();                  // buffer `cur` visible to all waves

        if (i + 1 < kIters)               // prefetch next A frag; latency hidden by WMMAs
            load_frag(aNxt, aRow + (size_t)(i + 1) * 32, lda);

#pragma unroll
        for (int nt = 0; nt < 4; ++nt) {
            Frag b;
            load_frag_lds(b, &tileB[cur][0], nt * 16);
            acc[nt] = __builtin_amdgcn_wmma_f32_16x16x32_bf16(
                false, aCur.v, false, b.v, (short)0, acc[nt], false, false);
        }
        __syncthreads();                  // reads of buffer `cur` done before rewrite
    };

    for (int i = 0; i < kIters; i += 2) { // explicit ping-pong: no frag copies
        kstep(i,     a0, a1);
        kstep(i + 1, a1, a0);
    }

    if (mode == 0) {
        __bf16* Cb = (__bf16*)Cout + (size_t)bz * sC;
#pragma unroll
        for (int nt = 0; nt < 4; ++nt)
#pragma unroll
            for (int r = 0; r < 8; ++r) {
                const int gm = m0 + mb + r, gn = n0 + nt * 16 + nl;
                Cb[(size_t)gm * ldc + gn] = f2bf(acc[nt][r]);
            }
    } else {
        float* Cf = (float*)Cout + (size_t)bz * sC;
#pragma unroll
        for (int nt = 0; nt < 4; ++nt)
#pragma unroll
            for (int r = 0; r < 8; ++r) {
                const int gm = m0 + mb + r, gn = n0 + nt * 16 + nl;
                float v = acc[nt][r] * scale;
                if (bias) v += bias[gn];
                if (causal && gn > gm) v = NEG_INF;
                Cf[(size_t)gm * ldc + gn] = v;
            }
    }
}

// ---------------- row softmax over S[row][0..T), write bf16 P ----------------
__global__ void softmax_row_k(const float* __restrict__ S, __bf16* __restrict__ P) {
    __shared__ float red[256];
    const int row = blockIdx.x;
    const int tid = threadIdx.x;
    const float* r = S + (size_t)row * TT;

    float vals[TT / 256];
    float m = NEG_INF;
#pragma unroll
    for (int i = 0; i < TT / 256; ++i) {
        vals[i] = r[tid + i * 256];
        m = fmaxf(m, vals[i]);
    }
    red[tid] = m; __syncthreads();
    for (int s = 128; s > 0; s >>= 1) {
        if (tid < s) red[tid] = fmaxf(red[tid], red[tid + s]);
        __syncthreads();
    }
    m = red[0]; __syncthreads();

    float sum = 0.f;
#pragma unroll
    for (int i = 0; i < TT / 256; ++i) {
        vals[i] = __expf(vals[i] - m);
        sum += vals[i];
    }
    red[tid] = sum; __syncthreads();
    for (int s = 128; s > 0; s >>= 1) {
        if (tid < s) red[tid] += red[tid + s];
        __syncthreads();
    }
    const float inv = 1.f / red[0];

    __bf16* p = P + (size_t)row * TT;
#pragma unroll
    for (int i = 0; i < TT / 256; ++i)
        p[tid + i * 256] = f2bf(vals[i] * inv);
}

// ---------------- host-side launch ----------------
extern "C" void kernel_launch(void* const* d_in, const int* in_sizes, int n_in,
                              void* d_out, int out_size, void* d_ws, size_t ws_size,
                              hipStream_t stream) {
    (void)in_sizes; (void)n_in; (void)out_size; (void)ws_size;

    const float* x     = (const float*)d_in[0];   // [B,T,C]
    const float* qkv_w = (const float*)d_in[1];   // [3C,C]
    const float* out_w = (const float*)d_in[2];   // [C,C]
    const float* out_b = (const float*)d_in[3];   // [C]
    float* y = (float*)d_out;                     // [B,T,C]

    char* ws = (char*)d_ws;
    __bf16* xb    = (__bf16*)(ws);                  // 16.8 MB  [B*T, C]
    __bf16* wqkvb = (__bf16*)(ws + 16777216);       //  6.3 MB  [3C, C]
    __bf16* woutb = (__bf16*)(ws + 23068672);       //  2.1 MB  [C, C]
    __bf16* qkvb  = (__bf16*)(ws + 25165824);       // 50.3 MB  [B,T,3C]
    __bf16* vt    = (__bf16*)(ws + 75497472);       // 16.8 MB  [B,C,T]
    float*  Sb    = (float* )(ws + 92274688);       // 67.1 MB  [B,T,T]
    __bf16* Pb    = (__bf16*)(ws + 159383552);      // 33.6 MB  [B,T,T]
    __bf16* ob    = (__bf16*)(ws + 192937984);      // 16.8 MB  [B,T,C]

    const int nX = BB * TT * CC, nWq = 3 * CC * CC, nWo = CC * CC;
    cvt_f32_to_bf16_k<<<(nX  / 4 + 255) / 256, 256, 0, stream>>>(x,     xb,    nX);
    cvt_f32_to_bf16_k<<<(nWq / 4 + 255) / 256, 256, 0, stream>>>(qkv_w, wqkvb, nWq);
    cvt_f32_to_bf16_k<<<(nWo / 4 + 255) / 256, 256, 0, stream>>>(out_w, woutb, nWo);

    // QKV projection: [8192,1024] x [3072,1024]^T -> bf16 [8192,3072]
    gemm_bf16_wmma_k<<<dim3(BB * TT / 128, 3 * CC / 64, 1), 256, 0, stream>>>(
        xb, wqkvb, qkvb, BB * TT, 3 * CC, CC, CC, CC, 3 * CC,
        0, 1.f, nullptr, 0, 0, 0, 0);

    // V -> Vt [B, C, T]
    transpose_v_k<<<dim3(TT / 32, CC / 32, BB), dim3(32, 32), 0, stream>>>(qkvb, vt);

    // S = (Q K^T)/sqrt(C) + causal mask, f32 [B,T,T]
    gemm_bf16_wmma_k<<<dim3(TT / 128, TT / 64, BB), 256, 0, stream>>>(
        qkvb, qkvb + CC, Sb, TT, TT, CC, 3 * CC, 3 * CC, TT,
        1, 0.03125f, nullptr,
        (long long)TT * 3 * CC, (long long)TT * 3 * CC, (long long)TT * TT, 1);

    // P = softmax(S), bf16
    softmax_row_k<<<BB * TT, 256, 0, stream>>>(Sb, Pb);

    // O = P @ V : A=P[T,T], Bt=Vt[C,T] -> bf16 [B,T,C]
    gemm_bf16_wmma_k<<<dim3(TT / 128, CC / 64, BB), 256, 0, stream>>>(
        Pb, vt, ob, TT, CC, TT, TT, TT, CC,
        0, 1.f, nullptr,
        (long long)TT * TT, (long long)CC * TT, (long long)TT * CC, 0);

    // Y = O @ out_w^T + out_b, f32 to d_out
    gemm_bf16_wmma_k<<<dim3(BB * TT / 128, CC / 64, 1), 256, 0, stream>>>(
        ob, woutb, y, BB * TT, CC, CC, CC, CC, CC,
        1, 1.f, out_b, 0, 0, 0, 0);
}
